// MotionEncoder_18348100288714
// MI455X (gfx1250) — matrix operand order repaired and checked by reading, not compiled, not run on hardware
//
#include <hip/hip_runtime.h>
#include <hip/hip_bf16.h>

// ---------------------------------------------------------------------------
// MotionEncoder for MI455X (gfx1250): bf16 WMMA path.
//   B=64, F=120, NJ=17, D=1024, IN_DIM=37, OUT=51, NL=2
// All GEMMs use v_wmma_f32_16x16x32_bf16. Weights converted once to bf16,
// transposed to [N][Kpad] so B fragments are contiguous 16B loads.
// LSTM layer = single fused kernel: one wave owns the 4 gate tiles (i,f,g,o)
// of one 16x16 hidden tile -> A fragment reused 4x per k-step, cell update
// fused in-register (no gates round-trip).
// ---------------------------------------------------------------------------

typedef __bf16 bf16;
typedef __attribute__((ext_vector_type(16))) __bf16 v16bf;
typedef __attribute__((ext_vector_type(8)))  __bf16 v8bf;
typedef __attribute__((ext_vector_type(8)))  float  v8f;

#define BB   64
#define FF   120
#define DD   1024
#define OUTD 51
#define XCW  1088   // 1024 (xe) + 51 (pred) + padding -> multiple of 32

// ---- fragment loaders (contiguous k order shared by A and B, wave32) --------------

// A: 16x32 bf16 tile from row-major activation matrix (row stride = lda elems)
// lane 0-15  : row M=lane,    K = 0..15
// lane 16-31 : row M=lane-16, K = 16..31
__device__ __forceinline__ v16bf load_a_frag(const bf16* __restrict__ A, int lda,
                                             int mBase, int kBase, int lane) {
    int row = mBase + (lane & 15);
    int kh  = (lane >> 4) << 4;
    const bf16* p = A + (size_t)row * (size_t)lda + (size_t)(kBase + kh);
    v8bf lo = *(const v8bf*)p;
    v8bf hi = *(const v8bf*)(p + 8);
    v16bf r;
#pragma unroll
    for (int i = 0; i < 8; ++i) { r[i] = lo[i]; r[8 + i] = hi[i]; }
    return r;
}

// B: 32x16 bf16 tile from pre-transposed weights Wt[N][ldk]
// lane 0-15  : col N=lane,    K = 0..15
// lane 16-31 : col N=lane-16, K = 16..31
__device__ __forceinline__ v16bf load_b_frag(const bf16* __restrict__ Wt, int ldk,
                                             int nBase, int kBase, int lane) {
    int col = nBase + (lane & 15);
    int kh  = (lane >> 4) << 4;
    const bf16* p = Wt + (size_t)col * (size_t)ldk + (size_t)(kBase + kh);
    v8bf lo = *(const v8bf*)p;
    v8bf hi = *(const v8bf*)(p + 8);
    v16bf r;
#pragma unroll
    for (int i = 0; i < 8; ++i) { r[i] = lo[i]; r[8 + i] = hi[i]; }
    return r;
}

__device__ __forceinline__ v8f wmma_bf16(v16bf a, v16bf b, v8f c) {
    return __builtin_amdgcn_wmma_f32_16x16x32_bf16(
        /*neg_a=*/false, a, /*neg_b=*/false, b,
        /*c_mod=*/(short)0, c, /*reuse_a=*/false, /*reuse_b=*/false);
}

__device__ __forceinline__ float sigm(float x) { return 1.0f / (1.0f + __expf(-x)); }

// ---- prep / conversion kernels --------------------------------------------

// f32 W[K][N] -> bf16 Wt[Npad][Kpad] (transposed, zero padded)
__global__ void k_transpose(const float* __restrict__ W, bf16* __restrict__ Wt,
                            int K, int N, int Kpad, int Npad) {
    int idx = blockIdx.x * blockDim.x + threadIdx.x;
    if (idx >= Npad * Kpad) return;
    int n = idx / Kpad, k = idx % Kpad;
    float v = (n < N && k < K) ? W[(size_t)k * N + n] : 0.0f;
    Wt[idx] = (bf16)v;
}

// masked input -> bf16 Ax[B*F][64]  (IN_DIM=37 padded to 64)
__global__ void k_prep_x(const float* __restrict__ x, const int* __restrict__ mask,
                         bf16* __restrict__ Ax) {
    int idx = blockIdx.x * blockDim.x + threadIdx.x;
    if (idx >= BB * FF * 64) return;
    int k = idx & 63, row = idx >> 6;
    float v = 0.0f;
    if (k < 37) {
        v = x[(size_t)row * 37 + k];
        if (k < 34 && mask[(size_t)row * 17 + (k >> 1)]) v = 0.0f;
    }
    Ax[idx] = (bf16)v;
}

// init[B][88] -> bf16 Ai[B][96] padded
__global__ void k_prep_init(const float* __restrict__ init, bf16* __restrict__ Ai) {
    int idx = blockIdx.x * blockDim.x + threadIdx.x;
    if (idx >= BB * 96) return;
    int b = idx / 96, k = idx % 96;
    float v = (k < 88) ? init[(size_t)b * 88 + k] : 0.0f;
    Ai[idx] = (bf16)v;
}

// zero pred/pad columns of xc for every (b,t); write pred0 (init[:,:51]) at t==0
__global__ void k_init_xc_pred(const float* __restrict__ init, bf16* __restrict__ xc) {
    int idx = blockIdx.x * blockDim.x + threadIdx.x;
    if (idx >= BB * FF * 64) return;
    int j = idx & 63, row = idx >> 6;       // row = b*120 + t
    int t = row % FF, b = row / FF;
    float v = 0.0f;
    if (t == 0 && j < OUTD) v = init[(size_t)b * 88 + j];
    xc[(size_t)row * XCW + 1024 + j] = (bf16)v;
}

// split init-MLP output o3[B][4096] into h0,h1 (bf16) and c0,c1 (f32)
__global__ void k_scatter_init(const float* __restrict__ o3,
                               bf16* __restrict__ h0bf, bf16* __restrict__ h1bf,
                               float* __restrict__ c0, float* __restrict__ c1) {
    int idx = blockIdx.x * blockDim.x + threadIdx.x;
    if (idx >= BB * 4096) return;
    int b = idx >> 12, col = idx & 4095;
    float v = o3[idx];
    int seg = col >> 10, d = col & 1023;
    size_t p = (size_t)b * DD + d;
    if      (seg == 0) h0bf[p] = (bf16)v;
    else if (seg == 1) h1bf[p] = (bf16)v;
    else if (seg == 2) c0[p] = v;
    else               c1[p] = v;
}

// ---- generic wave-tile GEMM: D = A(bf16) @ Wt(bf16) + bias ----------------
__global__ void k_gemm_wave(const bf16* __restrict__ A, int lda, int Kpad,
                            const bf16* __restrict__ Wt,
                            const float* __restrict__ bias,
                            int tilesM, int tilesN,
                            float* __restrict__ outF, int ldoF,
                            bf16* __restrict__ outB, int ldoB,
                            int relu) {
    int wid = (blockIdx.x * blockDim.x + threadIdx.x) >> 5;
    if (wid >= tilesM * tilesN) return;
    int lane = threadIdx.x & 31;
    int mT = wid % tilesM, nT = wid / tilesM;
    int mBase = mT * 16, nBase = nT * 16;
    int col = nBase + (lane & 15);
    float bv = bias ? bias[col] : 0.0f;
    v8f acc;
#pragma unroll
    for (int i = 0; i < 8; ++i) acc[i] = bv;
    for (int kb = 0; kb < Kpad; kb += 32) {
        v16bf a = load_a_frag(A, lda, mBase, kb, lane);
        v16bf b = load_b_frag(Wt, Kpad, nBase, kb, lane);
        acc = wmma_bf16(a, b, acc);
    }
    int rbase = mBase + ((lane >> 4) << 3);
#pragma unroll
    for (int r = 0; r < 8; ++r) {
        float v = acc[r];
        if (relu) v = v > 0.0f ? v : 0.0f;
        int row = rbase + r;
        if (outF) outF[(size_t)row * ldoF + col] = v;
        if (outB) outB[(size_t)row * ldoB + col] = (bf16)v;
    }
}

// ---- fused LSTM layer -----------------------------------------------------
// gates = A1@W1t + hprev@WhhT + bih + bhh ; then cell update, all in-register.
// One wave per 16x16 hidden tile (4 mT x 64 nT = 256 waves); each wave holds
// the 4 gate accumulators so the A fragment is reused 4x per k-step.
__global__ void k_lstm_fused(const bf16* __restrict__ A1, int lda1, int K1,
                             const bf16* __restrict__ W1t,      // [4096][K1]
                             const bf16* __restrict__ hprev,    // [64][1024]
                             const bf16* __restrict__ WhhT,     // [4096][1024]
                             const float* __restrict__ bih,
                             const float* __restrict__ bhh,
                             float* __restrict__ cst,           // [64][1024] in/out
                             bf16*  __restrict__ houtB,         // [64][1024]
                             float* __restrict__ ctxOut, int ldCtx) {
    int wid = (blockIdx.x * blockDim.x + threadIdx.x) >> 5;
    if (wid >= 4 * 64) return;
    int lane = threadIdx.x & 31;
    int mT = wid & 3, nT = wid >> 2;
    int mBase = mT * 16, nBase = nT * 16;
    int col = nBase + (lane & 15);

    v8f acc[4];
#pragma unroll
    for (int g = 0; g < 4; ++g) {
        float bv = bih[g * DD + col] + bhh[g * DD + col];
#pragma unroll
        for (int i = 0; i < 8; ++i) acc[g][i] = bv;
    }
    for (int kb = 0; kb < K1; kb += 32) {
        v16bf a = load_a_frag(A1, lda1, mBase, kb, lane);
#pragma unroll
        for (int g = 0; g < 4; ++g) {
            v16bf b = load_b_frag(W1t, K1, g * DD + nBase, kb, lane);
            acc[g] = wmma_bf16(a, b, acc[g]);
        }
    }
    for (int kb = 0; kb < DD; kb += 32) {
        v16bf a = load_a_frag(hprev, DD, mBase, kb, lane);
#pragma unroll
        for (int g = 0; g < 4; ++g) {
            v16bf b = load_b_frag(WhhT, DD, g * DD + nBase, kb, lane);
            acc[g] = wmma_bf16(a, b, acc[g]);
        }
    }
    int rbase = mBase + ((lane >> 4) << 3);
#pragma unroll
    for (int r = 0; r < 8; ++r) {
        int row = rbase + r;
        size_t p = (size_t)row * DD + col;
        float ig = sigm(acc[0][r]);
        float fg = sigm(acc[1][r]);
        float gg = tanhf(acc[2][r]);
        float og = sigm(acc[3][r]);
        float cn = fg * cst[p] + ig * gg;
        float hn = og * tanhf(cn);
        cst[p]   = cn;
        houtB[p] = (bf16)hn;
        if (ctxOut) ctxOut[(size_t)row * ldCtx + col] = hn;
    }
}

// ---- decoder: pred = h1 @ dec_Wt + dec_b; writes preds, ctx tail, next xc --
__global__ void k_dec(const bf16* __restrict__ h1bf,
                      const bf16* __restrict__ decWt,   // [64][1024]
                      const float* __restrict__ dec_b,
                      float* __restrict__ dOut, bf16* __restrict__ xc, int t) {
    int wid = threadIdx.x >> 5;          // blockDim = 512 -> 16 waves
    if (wid >= 16) return;
    int lane = threadIdx.x & 31;
    int mT = wid & 3, nT = wid >> 2;
    int mBase = mT * 16, nBase = nT * 16;
    int col = nBase + (lane & 15);
    float bv = (col < OUTD) ? dec_b[col] : 0.0f;
    v8f acc;
#pragma unroll
    for (int i = 0; i < 8; ++i) acc[i] = bv;
    for (int kb = 0; kb < DD; kb += 32) {
        v16bf a = load_a_frag(h1bf, DD, mBase, kb, lane);
        v16bf b = load_b_frag(decWt, DD, nBase, kb, lane);
        acc = wmma_bf16(a, b, acc);
    }
    if (col < OUTD) {
        int rbase = mBase + ((lane >> 4) << 3);
        float* preds = dOut;
        float* ctx   = dOut + (size_t)BB * FF * OUTD;
#pragma unroll
        for (int r = 0; r < 8; ++r) {
            int bi = rbase + r;
            float v = acc[r];
            size_t rt = (size_t)bi * FF + t;
            preds[rt * OUTD + col] = v;
            ctx[rt * (DD + OUTD) + DD + col] = v;
            if (t < FF - 1)
                xc[(rt + 1) * XCW + 1024 + col] = (bf16)v;
        }
    }
}

// ---------------------------------------------------------------------------
extern "C" void kernel_launch(void* const* d_in, const int* in_sizes, int n_in,
                              void* d_out, int out_size, void* d_ws, size_t ws_size,
                              hipStream_t stream) {
    const float* x       = (const float*)d_in[0];
    const float* init    = (const float*)d_in[1];
    const int*   mask    = (const int*)d_in[2];
    const float* embed_W = (const float*)d_in[3];
    const float* embed_b = (const float*)d_in[4];
    const float* ni_W1   = (const float*)d_in[5];
    const float* ni_b1   = (const float*)d_in[6];
    const float* ni_W2   = (const float*)d_in[7];
    const float* ni_b2   = (const float*)d_in[8];
    const float* ni_W3   = (const float*)d_in[9];
    const float* ni_b3   = (const float*)d_in[10];
    const float* Wih0    = (const float*)d_in[11];
    const float* Whh0    = (const float*)d_in[12];
    const float* bih0    = (const float*)d_in[13];
    const float* bhh0    = (const float*)d_in[14];
    const float* Wih1    = (const float*)d_in[15];
    const float* Whh1    = (const float*)d_in[16];
    const float* bih1    = (const float*)d_in[17];
    const float* bhh1    = (const float*)d_in[18];
    const float* dec_W   = (const float*)d_in[19];
    const float* dec_b   = (const float*)d_in[20];
    float* out = (float*)d_out;
    (void)in_sizes; (void)n_in; (void)out_size; (void)ws_size;

    char* base = (char*)d_ws;
    size_t off = 0;
    auto alloc = [&](size_t bytes) -> void* {
        void* p = base + off;
        off += (bytes + 255) & ~(size_t)255;
        return p;
    };

    bf16*  Ax    = (bf16*)alloc((size_t)BB * FF * 64 * 2);
    bf16*  Ai    = (bf16*)alloc((size_t)BB * 96 * 2);
    bf16*  xc    = (bf16*)alloc((size_t)BB * FF * XCW * 2);
    bf16*  eWt   = (bf16*)alloc((size_t)1024 * 64 * 2);
    bf16*  W1t   = (bf16*)alloc((size_t)1024 * 96 * 2);
    bf16*  W2t   = (bf16*)alloc((size_t)2048 * 1024 * 2);
    bf16*  W3t   = (bf16*)alloc((size_t)4096 * 2048 * 2);
    bf16*  Wih0t = (bf16*)alloc((size_t)4096 * XCW * 2);
    bf16*  Whh0t = (bf16*)alloc((size_t)4096 * 1024 * 2);
    bf16*  Wih1t = (bf16*)alloc((size_t)4096 * 1024 * 2);
    bf16*  Whh1t = (bf16*)alloc((size_t)4096 * 1024 * 2);
    bf16*  dWt   = (bf16*)alloc((size_t)64 * 1024 * 2);
    bf16*  t1    = (bf16*)alloc((size_t)BB * 1024 * 2);
    bf16*  t2    = (bf16*)alloc((size_t)BB * 2048 * 2);
    float* o3    = (float*)alloc((size_t)BB * 4096 * 4);
    bf16*  h0bf  = (bf16*)alloc((size_t)2 * BB * DD * 2);
    bf16*  h1bf  = (bf16*)alloc((size_t)2 * BB * DD * 2);
    float* c0    = (float*)alloc((size_t)BB * DD * 4);
    float* c1    = (float*)alloc((size_t)BB * DD * 4);

    auto grd = [](size_t n, int blk) { return (unsigned)((n + blk - 1) / blk); };

    // --- weight conversion / transpose (once per launch) ---
    k_transpose<<<grd(1024 * 64, 256), 256, 0, stream>>>(embed_W, eWt, 37, 1024, 64, 1024);
    k_transpose<<<grd(1024 * 96, 256), 256, 0, stream>>>(ni_W1, W1t, 88, 1024, 96, 1024);
    k_transpose<<<grd((size_t)2048 * 1024, 256), 256, 0, stream>>>(ni_W2, W2t, 1024, 2048, 1024, 2048);
    k_transpose<<<grd((size_t)4096 * 2048, 256), 256, 0, stream>>>(ni_W3, W3t, 2048, 4096, 2048, 4096);
    k_transpose<<<grd((size_t)4096 * XCW, 256), 256, 0, stream>>>(Wih0, Wih0t, 1075, 4096, XCW, 4096);
    k_transpose<<<grd((size_t)4096 * 1024, 256), 256, 0, stream>>>(Whh0, Whh0t, 1024, 4096, 1024, 4096);
    k_transpose<<<grd((size_t)4096 * 1024, 256), 256, 0, stream>>>(Wih1, Wih1t, 1024, 4096, 1024, 4096);
    k_transpose<<<grd((size_t)4096 * 1024, 256), 256, 0, stream>>>(Whh1, Whh1t, 1024, 4096, 1024, 4096);
    k_transpose<<<grd(64 * 1024, 256), 256, 0, stream>>>(dec_W, dWt, 1024, OUTD, 1024, 64);

    // --- input prep ---
    k_prep_x<<<grd((size_t)BB * FF * 64, 256), 256, 0, stream>>>(x, mask, Ax);
    k_prep_init<<<grd(BB * 96, 256), 256, 0, stream>>>(init, Ai);
    k_init_xc_pred<<<grd((size_t)BB * FF * 64, 256), 256, 0, stream>>>(init, xc);

    // --- embed: xe = maskedX @ embed_W + b -> xc[:, :, 0:1024] (bf16) ---
    k_gemm_wave<<<(480 * 64) / 8, 256, 0, stream>>>(Ax, 64, 64, eWt, embed_b,
                                                    480, 64, nullptr, 0, xc, XCW, 0);

    // --- init MLP -> h0,h1,c0,c1 ---
    k_gemm_wave<<<(4 * 64) / 8, 256, 0, stream>>>(Ai, 96, 96, W1t, ni_b1,
                                                  4, 64, nullptr, 0, t1, 1024, 1);
    k_gemm_wave<<<(4 * 128) / 8, 256, 0, stream>>>(t1, 1024, 1024, W2t, ni_b2,
                                                   4, 128, nullptr, 0, t2, 2048, 1);
    k_gemm_wave<<<(4 * 256) / 8, 256, 0, stream>>>(t2, 2048, 2048, W3t, ni_b3,
                                                   4, 256, o3, 4096, nullptr, 0, 0);
    k_scatter_init<<<grd((size_t)BB * 4096, 256), 256, 0, stream>>>(o3, h0bf, h1bf, c0, c1);

    // --- LSTM scan: 120 sequential steps, 3 kernels each ---
    const size_t HSZ = (size_t)BB * DD;
    float* ctxBase = out + (size_t)BB * FF * OUTD;
    for (int t = 0; t < FF; ++t) {
        const bf16* h0in  = h0bf + (size_t)(t & 1) * HSZ;
        bf16*       h0out = h0bf + (size_t)((t + 1) & 1) * HSZ;
        const bf16* h1in  = h1bf + (size_t)(t & 1) * HSZ;
        bf16*       h1out = h1bf + (size_t)((t + 1) & 1) * HSZ;

        // layer 0: A1 = [xe_t | pred_t] (row stride F*XCW), A2 = h0
        k_lstm_fused<<<32, 256, 0, stream>>>(xc + (size_t)t * XCW, FF * XCW, XCW, Wih0t,
                                             h0in, Whh0t, bih0, bhh0,
                                             c0, h0out, nullptr, 0);
        // layer 1: A1 = h0_new, A2 = h1; emits ctx (h1 f32) into motion_context
        k_lstm_fused<<<32, 256, 0, stream>>>(h0out, DD, DD, Wih1t,
                                             h1in, Whh1t, bih1, bhh1,
                                             c1, h1out,
                                             ctxBase + (size_t)t * (DD + OUTD),
                                             FF * (DD + OUTD));
        // decoder: pred -> d_out preds, ctx tail, next step's concat input
        k_dec<<<1, 512, 0, stream>>>(h1out, dWt, dec_b, out, xc, t);
    }
}
